// TGNSimpleModel_32547262169245
// MI455X (gfx1250) — compile-verified, matrix operand order checked by reading, not run on hardware
//
#include <hip/hip_runtime.h>
#include <hip/hip_bf16.h>

// ---------------------------------------------------------------------------
// TGN step for MI455X (gfx1250, wave32, WMMA).
// Compute-bound (~36.6 GFLOP vs ~135MB gather traffic): bf16 WMMA with f32
// accumulation. Weights are converted ONCE per persistent block into LDS in
// pre-swizzled B-fragment order; blocks then grid-stride over 128-row tiles
// (8 wave32s x 16 rows) so weight prep is amortized instead of repeated for
// every tile. Gather is float4-vectorized; next tile's indices prefetched
// with global_prefetch_b8.
// ---------------------------------------------------------------------------

typedef __attribute__((ext_vector_type(16))) __bf16 v16bf;
typedef __attribute__((ext_vector_type(8)))  __bf16 v8bf;
typedef __attribute__((ext_vector_type(4)))  __bf16 v4bf;
typedef __attribute__((ext_vector_type(8)))  float  v8f;
typedef __attribute__((ext_vector_type(4)))  float  f32x4;

#define ROWS    128
#define THREADS 256
#define PBLOCKS 1024  // persistent blocks; tiles assigned grid-stride
#define XLD     136   // 128 cols + 8 pad (row stride 272B -> bank staggered, 16B aligned)
#define HLD     72    // 64 cols + 8 pad  (row stride 144B, 16B aligned)

// LDS element counts (bf16)
#define N_W1   8192   // msg_w1  64x128
#define N_W2   4096   // msg_w2  64x64
#define N_WIH 12288   // gru_w_ih 192x64
#define N_WHH 12288   // gru_w_hh 192x64
#define N_WLP  8192   // lp_w1   64x128
#define N_X   (ROWS*XLD)
#define N_HID (ROWS*HLD)
#define N_MSG (ROWS*HLD)
#define SMEM_BYTES ((N_W1+N_W2+N_WIH+N_WHH+N_WLP+N_X+N_HID+N_MSG)*2 + 640*4)

__device__ __forceinline__ float sigmoidf_(float x) {
  return 1.0f / (1.0f + __expf(-x));
}
__device__ __forceinline__ float tanhf_(float x) {
  float e = __expf(-2.0f * x);
  return (1.0f - e) / (1.0f + e);
}

__device__ __forceinline__ v8f wmma_bf16(v16bf a, v16bf b, v8f c) {
  return __builtin_amdgcn_wmma_f32_16x16x32_bf16(
      /*neg_a=*/false, a, /*neg_b=*/false, b,
      /*c_mod=*/(short)0, c, /*reuse_a=*/false, /*reuse_b=*/false);
}

// A fragment: 16x32 bf16 tile, rows row0..row0+15 of a padded row-major LDS
// buffer. Lane m = l&15 carries row m; lane>>4 selects the K half.
// Each lane loads 32 contiguous bytes (two 16B-aligned v8bf loads).
__device__ __forceinline__ v16bf fragA(const __bf16* base, int ld, int row0, int kbase) {
  int l = threadIdx.x & 31;
  const __bf16* p = base + (row0 + (l & 15)) * ld + kbase + ((l >> 4) << 4);
  union { v16bf v; v8bf h[2]; } u;
  u.h[0] = *(const v8bf*)(p);
  u.h[1] = *(const v8bf*)(p + 8);
  return u.v;
}

// B fragment: pre-swizzled in LDS so lane l reads 32 contiguous bytes at
// frag*1024 + l*32.
__device__ __forceinline__ v16bf fragB(const __bf16* w, int frag) {
  int l = threadIdx.x & 31;
  const __bf16* p = w + (frag << 9) + (l << 4);
  union { v16bf v; v8bf h[2]; } u;
  u.h[0] = *(const v8bf*)(p);
  u.h[1] = *(const v8bf*)(p + 8);
  return u.v;
}

// Convert an f32 weight matrix W[Ntot][Ktot] (row-major, used as x @ W.T,
// i.e. B[k][n] = W[n][k]) into bf16 fragments in LDS.
// Fragment (nt,kt) covers n in [16nt,16nt+16), k in [32kt,32kt+32).
// Within a fragment: lane = (k_half<<4)|n_local, j = k&15.
__device__ __forceinline__ void prep_weights(const float* __restrict__ W,
                                             __bf16* dst, int Ntot, int Ktot) {
  int nKt = Ktot >> 5;
  int total = Ntot * Ktot;
  for (int e = threadIdx.x; e < total; e += THREADS) {
    int n = e / Ktot, k = e - n * Ktot;
    int nt = n >> 4, nl = n & 15;
    int kt = k >> 5, kl = k & 31;
    int lane = ((kl >> 4) << 4) | nl;
    int j = kl & 15;
    dst[(((nt * nKt + kt) << 9) + (lane << 4)) + j] = (__bf16)W[e];
  }
}

__global__ __launch_bounds__(THREADS)
void tgn_step_kernel(const int*   __restrict__ node_pairs,
                     const float* __restrict__ memory,
                     const float* __restrict__ msg_w1,
                     const float* __restrict__ msg_b1,
                     const float* __restrict__ msg_w2,
                     const float* __restrict__ msg_b2,
                     const float* __restrict__ gru_w_ih,
                     const float* __restrict__ gru_w_hh,
                     const float* __restrict__ gru_b_ih,
                     const float* __restrict__ gru_b_hh,
                     const float* __restrict__ lp_w1,
                     const float* __restrict__ lp_b1,
                     const float* __restrict__ lp_w2,
                     const float* __restrict__ lp_b2,
                     float* __restrict__ out,
                     int nrows, int ntiles) {
  extern __shared__ char smem_raw[];
  __bf16* W1s  = (__bf16*)smem_raw;
  __bf16* W2s  = W1s  + N_W1;
  __bf16* Wih  = W2s  + N_W2;
  __bf16* Whh  = Wih  + N_WIH;
  __bf16* Wlp1 = Whh  + N_WHH;
  __bf16* xs   = Wlp1 + N_WLP;   // 128 x 128 activations (n1||n2), later (n1'||n2')
  __bf16* hids = xs   + N_X;     // 128 x 64 msg hidden
  __bf16* msgs = hids + N_HID;   // 128 x 64 messages
  float*  b1s  = (float*)(msgs + N_MSG);
  float*  b2s   = b1s  + 64;
  float*  bihs  = b2s  + 64;
  float*  bhhs  = bihs + 192;
  float*  lpb1s = bhhs + 192;
  float*  lpw2s = lpb1s + 64;

  // ---- One-time per-block weight/bias prep (amortized over all tiles) ----
  prep_weights(msg_w1,   W1s,  64, 128);
  prep_weights(msg_w2,   W2s,  64,  64);
  prep_weights(gru_w_ih, Wih, 192,  64);
  prep_weights(gru_w_hh, Whh, 192,  64);
  prep_weights(lp_w1,    Wlp1, 64, 128);
  for (int i = threadIdx.x; i < 64; i += THREADS) {
    b1s[i]   = msg_b1[i];
    b2s[i]   = msg_b2[i];
    lpb1s[i] = lp_b1[i];
    lpw2s[i] = lp_w2[i];
  }
  for (int i = threadIdx.x; i < 192; i += THREADS) {
    bihs[i] = gru_b_ih[i];
    bhhs[i] = gru_b_hh[i];
  }

  const int l  = threadIdx.x & 31;
  const int ln = l & 15;          // N index within a tile
  const int lh = l >> 4;          // C-layout row half (M += 8)
  const int r0 = (threadIdx.x >> 5) << 4;  // local row base for this wave
  const float lb2 = lp_b2[0];

  for (int tile = blockIdx.x; tile < ntiles; tile += gridDim.x) {
    const int grow0 = tile * ROWS;

    __syncthreads();  // previous tile's compute done before overwriting xs
                      // (also orders weight prep before first compute)

    // Prefetch next tile's node indices into cache while we work.
    int ntile = tile + gridDim.x;
    if (ntile < ntiles)
      __builtin_prefetch(&node_pairs[(size_t)ntile * ROWS * 2], 0, 0);

    // ---- Gather n1_mem||n2_mem -> bf16 x tile (float4-vectorized) ----
    for (int e = threadIdx.x; e < ROWS * 32; e += THREADS) {
      int r = e >> 5, c4 = e & 31;             // c4: 16B column chunk (0..31)
      int grow = grow0 + r;
      int rr = (grow < nrows) ? grow : 0;
      int node = node_pairs[rr * 2 + (c4 >> 4)];
      f32x4 mv = ((const f32x4*)memory)[node * 16 + (c4 & 15)];
      v4bf o;
      o.x = (__bf16)mv.x; o.y = (__bf16)mv.y;
      o.z = (__bf16)mv.z; o.w = (__bf16)mv.w;
      *(v4bf*)(xs + r * XLD + c4 * 4) = o;
    }
    __syncthreads();

    // ---- Per-wave compute: 16 rows each, all WMMA (EXEC all-ones) ----

    // Stage 1: hid = relu(x @ msg_w1.T + b1)   [16x64], K=128
    #pragma unroll
    for (int nt = 0; nt < 4; ++nt) {
      v8f acc = {};
      #pragma unroll
      for (int kt = 0; kt < 4; ++kt)
        acc = wmma_bf16(fragA(xs, XLD, r0, kt * 32), fragB(W1s, nt * 4 + kt), acc);
      float b = b1s[nt * 16 + ln];
      int col = nt * 16 + ln;
      #pragma unroll
      for (int v = 0; v < 8; ++v) {
        float t = acc[v] + b;
        t = t > 0.0f ? t : 0.0f;
        hids[(r0 + v + 8 * lh) * HLD + col] = (__bf16)t;
      }
    }

    // Stage 2: msg = hid @ msg_w2.T + b2   [16x64], K=64
    #pragma unroll
    for (int nt = 0; nt < 4; ++nt) {
      v8f acc = {};
      #pragma unroll
      for (int kt = 0; kt < 2; ++kt)
        acc = wmma_bf16(fragA(hids, HLD, r0, kt * 32), fragB(W2s, nt * 2 + kt), acc);
      float b = b2s[nt * 16 + ln];
      int col = nt * 16 + ln;
      #pragma unroll
      for (int v = 0; v < 8; ++v)
        msgs[(r0 + v + 8 * lh) * HLD + col] = (__bf16)(acc[v] + b);
    }

    // Stage 3: two GRU cells. h(cell0)=x[:,0:64], h(cell1)=x[:,64:128].
    v8f upd[2][4];
    #pragma unroll
    for (int cell = 0; cell < 2; ++cell) {
      int hb = cell * 64;
      #pragma unroll
      for (int ct = 0; ct < 4; ++ct) {
        v8f ir = {}, iz = {}, inn = {}, hr = {}, hz = {}, hn = {};
        #pragma unroll
        for (int kt = 0; kt < 2; ++kt) {
          v16bf Am = fragA(msgs, HLD, r0, kt * 32);
          v16bf Ah = fragA(xs, XLD, r0, hb + kt * 32);
          ir  = wmma_bf16(Am, fragB(Wih, (ct)     * 2 + kt), ir);
          iz  = wmma_bf16(Am, fragB(Wih, (4 + ct) * 2 + kt), iz);
          inn = wmma_bf16(Am, fragB(Wih, (8 + ct) * 2 + kt), inn);
          hr  = wmma_bf16(Ah, fragB(Whh, (ct)     * 2 + kt), hr);
          hz  = wmma_bf16(Ah, fragB(Whh, (4 + ct) * 2 + kt), hz);
          hn  = wmma_bf16(Ah, fragB(Whh, (8 + ct) * 2 + kt), hn);
        }
        int col = ct * 16 + ln;
        float bir = bihs[col], biz = bihs[64 + col], bin = bihs[128 + col];
        float bhr = bhhs[col], bhz = bhhs[64 + col], bhn = bhhs[128 + col];
        v8f o;
        #pragma unroll
        for (int v = 0; v < 8; ++v) {
          int row = r0 + v + 8 * lh;
          float hprev = (float)xs[row * XLD + hb + col];
          float r = sigmoidf_(ir[v]  + bir + hr[v] + bhr);
          float z = sigmoidf_(iz[v]  + biz + hz[v] + bhz);
          float n = tanhf_(inn[v] + bin + r * (hn[v] + bhn));
          o[v] = (1.0f - z) * n + z * hprev;
        }
        upd[cell][ct] = o;
      }
    }
    // Overwrite x with combined = n1_upd || n2_upd (this wave's rows only).
    #pragma unroll
    for (int cell = 0; cell < 2; ++cell)
      #pragma unroll
      for (int ct = 0; ct < 4; ++ct) {
        int col = cell * 64 + ct * 16 + ln;
        #pragma unroll
        for (int v = 0; v < 8; ++v)
          xs[(r0 + v + 8 * lh) * XLD + col] = (__bf16)upd[cell][ct][v];
      }

    // Stage 4+5: h = relu(combined @ lp_w1.T + lp_b1); out = sigmoid(h.lp_w2 + b2)
    float tot[8];
    #pragma unroll
    for (int v = 0; v < 8; ++v) tot[v] = lb2;
    #pragma unroll
    for (int nt = 0; nt < 4; ++nt) {
      v8f acc = {};
      #pragma unroll
      for (int kt = 0; kt < 4; ++kt)
        acc = wmma_bf16(fragA(xs, XLD, r0, kt * 32), fragB(Wlp1, nt * 4 + kt), acc);
      float b  = lpb1s[nt * 16 + ln];
      float w2 = lpw2s[nt * 16 + ln];
      #pragma unroll
      for (int v = 0; v < 8; ++v) {
        float t = acc[v] + b;
        t = t > 0.0f ? t : 0.0f;
        float p = t * w2;
        // reduce across the 16 lanes holding this row's 16 columns
        p += __shfl_xor(p, 1, 16);
        p += __shfl_xor(p, 2, 16);
        p += __shfl_xor(p, 4, 16);
        p += __shfl_xor(p, 8, 16);
        tot[v] += p;
      }
    }
    if (ln == 0) {
      int growbase = grow0 + r0 + 8 * lh;
      #pragma unroll
      for (int v = 0; v < 8; ++v) {
        int grow = growbase + v;
        if (grow < nrows) out[grow] = sigmoidf_(tot[v]);
      }
    }
  }
}

extern "C" void kernel_launch(void* const* d_in, const int* in_sizes, int n_in,
                              void* d_out, int out_size, void* d_ws, size_t ws_size,
                              hipStream_t stream) {
  const int*   node_pairs = (const int*)  d_in[0];
  const float* memory     = (const float*)d_in[1];
  const float* msg_w1     = (const float*)d_in[2];
  const float* msg_b1     = (const float*)d_in[3];
  const float* msg_w2     = (const float*)d_in[4];
  const float* msg_b2     = (const float*)d_in[5];
  const float* gru_w_ih   = (const float*)d_in[6];
  const float* gru_w_hh   = (const float*)d_in[7];
  const float* gru_b_ih   = (const float*)d_in[8];
  const float* gru_b_hh   = (const float*)d_in[9];
  const float* lp_w1      = (const float*)d_in[10];
  const float* lp_b1      = (const float*)d_in[11];
  const float* lp_w2      = (const float*)d_in[12];
  const float* lp_b2      = (const float*)d_in[13];
  float* out = (float*)d_out;

  int nrows  = in_sizes[0] / 2;                 // BATCH
  int ntiles = (nrows + ROWS - 1) / ROWS;       // 2048 for BATCH=262144
  int blocks = ntiles < PBLOCKS ? ntiles : PBLOCKS;

  tgn_step_kernel<<<blocks, THREADS, SMEM_BYTES, stream>>>(
      node_pairs, memory, msg_w1, msg_b1, msg_w2, msg_b2,
      gru_w_ih, gru_w_hh, gru_b_ih, gru_b_hh,
      lp_w1, lp_b1, lp_w2, lp_b2, out, nrows, ntiles);
}
